// GKT_8280696946854
// MI455X (gfx1250) — matrix operand order repaired and checked by reading, not compiled, not run on hardware
//
#include <hip/hip_runtime.h>
#include <hip/hip_bf16.h>
#include <math.h>

// ---------------------------------------------------------------------------
// Problem constants (from reference): B=256, C=1024, H=64, E=64, K=2
// ---------------------------------------------------------------------------
#define BB   256
#define CC   1024
#define HH   64
#define MI   128          // H + E
#define NKG  2            // K graphs
#define WPB  4            // waves per block (wave32)

typedef __attribute__((ext_vector_type(16))) _Float16 v16h;
typedef __attribute__((ext_vector_type(8)))  float    v8f;

// Fragment table base indices (each fragment = 512 halves = 1 KB in d_ws)
#define FR_WS1 0     // 16 : [kt4][nt4]        (128x64)
#define FR_WS2 16    // 8  : [kt2][nt4]        (64x64)
#define FR_WN1 24    // 64 : [k2][kt8][nt4]    (256x64)
#define FR_WN2 88    // 16 : [k2][kt2][nt4]    (64x64)
#define FR_WE  104   // 8
#define FR_WA  112   // 8
#define FR_WIH 120   // 24 : [kt2][nt12]       (64x192)
#define FR_WHH 144   // 24
#define NFRAG  168
#define ADJ_OFF_BYTES (NFRAG * 512 * 2)   // 172032 B, adj = float[2][C] after frags

// ---------------------------------------------------------------------------
// helpers
// ---------------------------------------------------------------------------
__device__ __forceinline__ float sigm(float x) { return 1.0f / (1.0f + __expf(-x)); }
__device__ __forceinline__ float clampf(float v, float lo, float hi) { return fminf(fmaxf(v, lo), hi); }
__device__ __forceinline__ v8f vsplat(float x) { v8f r = {x, x, x, x, x, x, x, x}; return r; }

__device__ __forceinline__ v8f wmma16(v16h a, v16h b, v8f c) {
  // D = A(16x32 f16) * B(32x16 f16) + C(16x16 f32)
  return __builtin_amdgcn_wmma_f32_16x16x32_f16(false, a, false, b, (short)0, c, false, false);
}

// Load B fragment packed by pack_kernel: per-lane-contiguous 16 halves.
__device__ __forceinline__ v16h load_b(const _Float16* frags, int f, int lane) {
  return *(const v16h*)(frags + (((size_t)f) << 9) + (lane << 4));
}

// Build A fragment (16x32, f16) from a row-major f16 LDS tile (leading dim ld).
// CDNA5 layout: lane holds row M = lane&15; halves 0..7 -> K = kbase+koff+i,
// halves 8..15 -> K = kbase+koff+16+i, koff = (lane>=16)*8.
__device__ __forceinline__ v16h load_a(const _Float16* t, int ld, int kbase, int lane,
                                       float scale, float lo, float hi) {
  const int row  = lane & 15;
  const int koff = kbase + ((lane >> 4) << 3);
  const _Float16* p = t + row * ld + koff;
  v16h a;
#pragma unroll
  for (int i = 0; i < 8; ++i) {
    float v0 = (float)p[i] * scale;
    float v1 = (float)p[16 + i] * scale;
    a[i]     = (_Float16)clampf(v0, lo, hi);
    a[8 + i] = (_Float16)clampf(v1, lo, hi);
  }
  return a;
}

// Store C/D accumulator tile to row-major f16 LDS tile.
// Layout: VGPR j, lanes<16 -> M=j ; lanes>=16 -> M=j+8 ; N = nt*16 + lane&15.
__device__ __forceinline__ void store_cd(_Float16* t, int ld, int nt, int lane, v8f acc) {
  const int col   = nt * 16 + (lane & 15);
  const int rbase = (lane >> 4) << 3;
#pragma unroll
  for (int j = 0; j < 8; ++j) t[(rbase + j) * ld + col] = (_Float16)acc[j];
}

// ---------------------------------------------------------------------------
// Kernel 1: pack all weight matrices (fp32) into WMMA B-fragment layout (f16)
// ---------------------------------------------------------------------------
__global__ void pack_kernel(const float* Ws1, const float* Ws2, const float* Wn1,
                            const float* Wn2, const float* We, const float* Wa,
                            const float* Wih, const float* Whh, _Float16* frags) {
  int tid = blockIdx.x * blockDim.x + threadIdx.x;
  if (tid >= NFRAG * 512) return;
  int f = tid >> 9, w = tid & 511, lane = w >> 4, i = w & 15;
  const float* src; int ncol, kt, nt;
  if (f < FR_WS2)      { src = Ws1; ncol = 64;  int r = f;          kt = r >> 2; nt = r & 3; }
  else if (f < FR_WN1) { src = Ws2; ncol = 64;  int r = f - FR_WS2; kt = r >> 2; nt = r & 3; }
  else if (f < FR_WN2) { int r = f - FR_WN1; src = Wn1 + (size_t)(r >> 5) * 256 * 64;
                         r &= 31; ncol = 64; kt = r >> 2; nt = r & 3; }
  else if (f < FR_WE)  { int r = f - FR_WN2; src = Wn2 + (size_t)(r >> 3) * 64 * 64;
                         r &= 7;  ncol = 64; kt = r >> 2; nt = r & 3; }
  else if (f < FR_WA)  { src = We;  ncol = 64;  int r = f - FR_WE;  kt = r >> 2; nt = r & 3; }
  else if (f < FR_WIH) { src = Wa;  ncol = 64;  int r = f - FR_WA;  kt = r >> 2; nt = r & 3; }
  else if (f < FR_WHH) { src = Wih; ncol = 192; int r = f - FR_WIH; kt = r / 12; nt = r % 12; }
  else                 { src = Whh; ncol = 192; int r = f - FR_WHH; kt = r / 12; nt = r % 12; }
  // B layout: i<8 -> K = kt*32 + 8*(lane>=16) + i ; i>=8 -> K = kt*32 + 16 + 8*(lane>=16) + (i-8)
  int kk  = kt * 32 + ((lane >> 4) << 3) + (i & 7) + ((i >> 3) << 4);
  int col = nt * 16 + (lane & 15);
  frags[tid] = (_Float16)src[(size_t)kk * ncol + col];
}

// ---------------------------------------------------------------------------
// Kernel 2: adj[k][c] = clip( (a0 @ graphs[k])[c] / max(sum(a0),1), -5, 5 )
//           with a0 = qt_kc_one_hot[qt[0]]
// ---------------------------------------------------------------------------
__global__ void adj_kernel(const float* onehot, const int* qt, const float* graphs,
                           float* adj) {
  __shared__ float a0[CC];
  const int tid = threadIdx.x;
  const int k = blockIdx.y;
  const int c = blockIdx.x * 256 + tid;
  const float* row = onehot + (size_t)qt[0] * CC;
  for (int i = tid; i < CC; i += 256) a0[i] = row[i];
  __syncthreads();
  float s = 0.f, dot = 0.f;
  const float* g = graphs + (size_t)k * CC * CC + c;
  for (int d = 0; d < CC; ++d) {
    float a = a0[d];
    s += a;
    dot += a * g[(size_t)d * CC];
  }
  float denom = fmaxf(s, 1.0f);
  adj[k * CC + c] = clampf(dot / denom, -5.0f, 5.0f);
}

// ---------------------------------------------------------------------------
// Kernel 3: fully fused pipeline, one wave per 16-row tile (fixed b, 16 c's).
// ---------------------------------------------------------------------------
__global__ void __launch_bounds__(WPB * 32) fused_kernel(
    const int* __restrict__ qt, const float* __restrict__ ht,
    const float* __restrict__ onehot, const float* __restrict__ kc,
    const float* __restrict__ nw,
    const float* __restrict__ bs1, const float* __restrict__ bs2,
    const float* __restrict__ bn1, const float* __restrict__ bn2,
    const float* __restrict__ ea,  const float* __restrict__ be,
    const float* __restrict__ ba,  const float* __restrict__ bih,
    const float* __restrict__ bhh, const float* __restrict__ Wp,
    const float* __restrict__ bp,  const _Float16* __restrict__ frags,
    const float* __restrict__ adj, float* __restrict__ out) {

  __shared__ _Float16 s_tmp [WPB][16][128];  // [ht | kc]  (f16)
  __shared__ _Float16 s_hid [WPB][16][64];   // hidden activations
  __shared__ _Float16 s_self[WPB][16][64];   // self_feat
  __shared__ _Float16 s_nf  [WPB][16][64];   // neighbor accumulation
  __shared__ _Float16 s_m   [WPB][16][64];   // m_next
  __shared__ _Float16 s_res [WPB][16][64];   // res (GRU input)
  __shared__ float    s_scal[WPB][4][16];    // mask, ea, adj0, adj1 per row

  const int lane = threadIdx.x & 31;
  const int wv   = threadIdx.x >> 5;
  const int tile = blockIdx.x * WPB + wv;
  const int b    = tile >> 6;          // 64 tiles along C
  const int c0   = (tile & 63) << 4;

  _Float16* tmp  = &s_tmp [wv][0][0];
  _Float16* hid  = &s_hid [wv][0][0];
  _Float16* slf  = &s_self[wv][0][0];
  _Float16* snf  = &s_nf  [wv][0][0];
  _Float16* smx  = &s_m   [wv][0][0];
  _Float16* srs  = &s_res [wv][0][0];
  float*    scal = &s_scal[wv][0][0];

  const float BIG = 3.0e38f;
  const int   cb15 = lane & 15;
  const int   rbase = (lane >> 4) << 3;

  // ---- phase 0: stage tile ------------------------------------------------
  for (int idx = lane; idx < 16 * 64; idx += 32) {
    int r = idx >> 6, col = idx & 63;
    tmp[r * 128 + col]      = (_Float16)ht[((size_t)b * CC + c0 + r) * HH + col];
    tmp[r * 128 + 64 + col] = (_Float16)kc[(size_t)(c0 + r) * HH + col];
  }
  if (lane < 16) {
    int c = c0 + lane;
    scal[0 * 16 + lane] = onehot[(size_t)qt[b] * CC + c];  // mask
    scal[1 * 16 + lane] = ea[c];                           // g
    scal[2 * 16 + lane] = adj[c];                          // adj[0]
    scal[3 * 16 + lane] = adj[CC + c];                     // adj[1]
  }
  __syncthreads();

  const float mask_me = scal[cb15];   // scale for this lane's A-row

  // ---- phase 1: self MLP layer 1: (tmp*mask)(16x128) @ Ws1 + bs1, relu ----
  v16h aS[4];
#pragma unroll
  for (int kt = 0; kt < 4; ++kt)
    aS[kt] = load_a(tmp, 128, kt * 32, lane, mask_me, -BIG, BIG);
#pragma unroll
  for (int nt = 0; nt < 4; ++nt) {
    v8f acc = vsplat(bs1[nt * 16 + cb15]);
#pragma unroll
    for (int kt = 0; kt < 4; ++kt)
      acc = wmma16(aS[kt], load_b(frags, FR_WS1 + kt * 4 + nt, lane), acc);
#pragma unroll
    for (int j = 0; j < 8; ++j) acc[j] = fmaxf(acc[j], 0.0f);
    store_cd(hid, 64, nt, lane, acc);
  }
  __syncthreads();

  // ---- phase 2: self MLP layer 2, relu, clip +-10 -> self_feat ------------
  {
    v16h a0f = load_a(hid, 64, 0,  lane, 1.0f, -BIG, BIG);
    v16h a1f = load_a(hid, 64, 32, lane, 1.0f, -BIG, BIG);
#pragma unroll
    for (int nt = 0; nt < 4; ++nt) {
      v8f acc = vsplat(bs2[nt * 16 + cb15]);
      acc = wmma16(a0f, load_b(frags, FR_WS2 + nt,     lane), acc);
      acc = wmma16(a1f, load_b(frags, FR_WS2 + 4 + nt, lane), acc);
#pragma unroll
      for (int j = 0; j < 8; ++j) acc[j] = clampf(fmaxf(acc[j], 0.0f), -10.0f, 10.0f);
      store_cd(slf, 64, nt, lane, acc);
    }
  }
  __syncthreads();

  // ---- phase 3: neighbor nets (K=2) + combine -> m_next -------------------
  // neigh_ht = clip([tmp*mask | tmp], -5, 5) : 16x256
  v16h aN[8];
#pragma unroll
  for (int kt = 0; kt < 4; ++kt)
    aN[kt] = load_a(tmp, 128, kt * 32, lane, mask_me, -5.0f, 5.0f);
#pragma unroll
  for (int kt = 4; kt < 8; ++kt)
    aN[kt] = load_a(tmp, 128, (kt - 4) * 32, lane, 1.0f, -5.0f, 5.0f);

  const float wmix = clampf(nw[0], 0.1f, 0.9f);

  for (int k = 0; k < NKG; ++k) {
#pragma unroll
    for (int nt = 0; nt < 4; ++nt) {
      v8f acc = vsplat(bn1[k * 64 + nt * 16 + cb15]);
#pragma unroll
      for (int kt = 0; kt < 8; ++kt)
        acc = wmma16(aN[kt], load_b(frags, FR_WN1 + k * 32 + kt * 4 + nt, lane), acc);
#pragma unroll
      for (int j = 0; j < 8; ++j) acc[j] = fmaxf(acc[j], 0.0f);
      store_cd(hid, 64, nt, lane, acc);
    }
    __syncthreads();

    v16h a0f = load_a(hid, 64, 0,  lane, 1.0f, -BIG, BIG);
    v16h a1f = load_a(hid, 64, 32, lane, 1.0f, -BIG, BIG);
#pragma unroll
    for (int nt = 0; nt < 4; ++nt) {
      v8f acc = vsplat(bn2[k * 64 + nt * 16 + cb15]);
      acc = wmma16(a0f, load_b(frags, FR_WN2 + k * 8 + nt,     lane), acc);
      acc = wmma16(a1f, load_b(frags, FR_WN2 + k * 8 + 4 + nt, lane), acc);
      const int colb = nt * 16 + cb15;
#pragma unroll
      for (int j = 0; j < 8; ++j) {
        int   row = rbase + j;
        float v   = clampf(fmaxf(acc[j], 0.0f), -5.0f, 5.0f);      // nb[k], clipped
        float adk = scal[(2 + k) * 16 + row];
        if (k == 0) {
          snf[row * 64 + colb] = (_Float16)clampf(adk * v, -5.0f, 5.0f);
        } else {
          float t  = clampf(wmix * (float)snf[row * 64 + colb]
                            + (1.0f - wmix) * adk * v, -5.0f, 5.0f);
          float mr = scal[row];
          float sf = (float)slf[row * 64 + colb];
          float mn = clampf(mr > 0.5f ? sf : t, -50.0f, 50.0f);
          smx[row * 64 + colb] = (_Float16)mn;
        }
      }
    }
    __syncthreads();
  }

  // ---- phase 4: res = m - g*sigmoid(m@We+be)*m + g*tanh(m@Wa+ba) ----------
  {
    v16h a0f = load_a(smx, 64, 0,  lane, 1.0f, -BIG, BIG);
    v16h a1f = load_a(smx, 64, 32, lane, 1.0f, -BIG, BIG);
#pragma unroll
    for (int nt = 0; nt < 4; ++nt) {
      const int colb = nt * 16 + cb15;
      v8f ae = vsplat(be[colb]);
      v8f aa = vsplat(ba[colb]);
      ae = wmma16(a0f, load_b(frags, FR_WE + nt,     lane), ae);
      ae = wmma16(a1f, load_b(frags, FR_WE + 4 + nt, lane), ae);
      aa = wmma16(a0f, load_b(frags, FR_WA + nt,     lane), aa);
      aa = wmma16(a1f, load_b(frags, FR_WA + 4 + nt, lane), aa);
#pragma unroll
      for (int j = 0; j < 8; ++j) {
        int   row = rbase + j;
        float m   = (float)smx[row * 64 + colb];
        float g   = scal[16 + row];
        float r   = m - g * sigm(ae[j]) * m + g * tanhf(aa[j]);
        srs[row * 64 + colb] = (_Float16)r;
      }
    }
  }
  __syncthreads();

  // ---- phase 5: GRU + output projection -----------------------------------
  v16h aR[2], aH[2];
  aR[0] = load_a(srs, 64, 0,  lane, 1.0f, -BIG, BIG);
  aR[1] = load_a(srs, 64, 32, lane, 1.0f, -BIG, BIG);
  aH[0] = load_a(tmp, 128, 0,  lane, 1.0f, -BIG, BIG);   // ht cols 0..63
  aH[1] = load_a(tmp, 128, 32, lane, 1.0f, -BIG, BIG);

  v8f y8 = vsplat(0.0f);
#pragma unroll
  for (int nt = 0; nt < 4; ++nt) {
    const int colb = nt * 16 + cb15;
    v8f ir = vsplat(bih[colb]);
    v8f iz = vsplat(bih[64 + colb]);
    v8f in = vsplat(bih[128 + colb]);
    v8f hr = vsplat(bhh[colb]);
    v8f hz = vsplat(bhh[64 + colb]);
    v8f hn = vsplat(bhh[128 + colb]);
#pragma unroll
    for (int kt = 0; kt < 2; ++kt) {
      ir = wmma16(aR[kt], load_b(frags, FR_WIH + kt * 12 + nt,     lane), ir);
      iz = wmma16(aR[kt], load_b(frags, FR_WIH + kt * 12 + 4 + nt, lane), iz);
      in = wmma16(aR[kt], load_b(frags, FR_WIH + kt * 12 + 8 + nt, lane), in);
      hr = wmma16(aH[kt], load_b(frags, FR_WHH + kt * 12 + nt,     lane), hr);
      hz = wmma16(aH[kt], load_b(frags, FR_WHH + kt * 12 + 4 + nt, lane), hz);
      hn = wmma16(aH[kt], load_b(frags, FR_WHH + kt * 12 + 8 + nt, lane), hn);
    }
    const float wp = Wp[colb];
#pragma unroll
    for (int j = 0; j < 8; ++j) {
      int   row = rbase + j;
      float r   = sigm(ir[j] + hr[j]);
      float z   = sigm(iz[j] + hz[j]);
      float n   = tanhf(in[j] + r * hn[j]);
      float hv  = ht[((size_t)b * CC + c0 + row) * HH + colb];   // fp32 ht (L2-hot)
      float hnx = (1.0f - z) * n + z * hv;
      y8[j] += hnx * wp;
    }
  }

  // reduce y over the 16 lanes holding the same row (N dimension)
#pragma unroll
  for (int j = 0; j < 8; ++j) {
    float v = y8[j];
    v += __shfl_xor(v, 1, 16);
    v += __shfl_xor(v, 2, 16);
    v += __shfl_xor(v, 4, 16);
    v += __shfl_xor(v, 8, 16);
    y8[j] = v;
  }
  if (cb15 == 0) {
    const float bpv = bp[0];
#pragma unroll
    for (int j = 0; j < 8; ++j) {
      int row = rbase + j;
      out[(size_t)b * CC + c0 + row] = 1.0f / (1.0f + __expf(-(y8[j] + bpv)));
    }
  }
}

// ---------------------------------------------------------------------------
// launch
// ---------------------------------------------------------------------------
extern "C" void kernel_launch(void* const* d_in, const int* in_sizes, int n_in,
                              void* d_out, int out_size, void* d_ws, size_t ws_size,
                              hipStream_t stream) {
  // input order from setup_inputs()
  const int*   qt     = (const int*)  d_in[1];
  const float* ht     = (const float*)d_in[2];
  const float* onehot = (const float*)d_in[3];
  const float* kc     = (const float*)d_in[4];
  const float* graphs = (const float*)d_in[5];
  const float* nw     = (const float*)d_in[6];
  const float* Ws1    = (const float*)d_in[7];
  const float* bs1    = (const float*)d_in[8];
  const float* Ws2    = (const float*)d_in[9];
  const float* bs2    = (const float*)d_in[10];
  const float* Wn1    = (const float*)d_in[11];
  const float* bn1    = (const float*)d_in[12];
  const float* Wn2    = (const float*)d_in[13];
  const float* bn2    = (const float*)d_in[14];
  const float* ea     = (const float*)d_in[15];
  const float* We     = (const float*)d_in[16];
  const float* be     = (const float*)d_in[17];
  const float* Wa     = (const float*)d_in[18];
  const float* ba     = (const float*)d_in[19];
  const float* Wih    = (const float*)d_in[20];
  const float* bih    = (const float*)d_in[21];
  const float* Whh    = (const float*)d_in[22];
  const float* bhh    = (const float*)d_in[23];
  const float* Wp     = (const float*)d_in[24];
  const float* bp     = (const float*)d_in[25];

  _Float16* frags = (_Float16*)d_ws;
  float*    adj   = (float*)((char*)d_ws + ADJ_OFF_BYTES);
  float*    out   = (float*)d_out;

  // 1) pack weights into WMMA B-fragment layout (f16)
  pack_kernel<<<(NFRAG * 512 + 255) / 256, 256, 0, stream>>>(
      Ws1, Ws2, Wn1, Wn2, We, Wa, Wih, Whh, frags);

  // 2) adjacency vector from batch-element-0 mask
  adj_kernel<<<dim3(CC / 256, NKG), 256, 0, stream>>>(onehot, qt, graphs, adj);

  // 3) fused pipeline: 16384 tiles, 4 waves/block
  const int tiles = BB * (CC / 16);
  fused_kernel<<<tiles / WPB, WPB * 32, 0, stream>>>(
      qt, ht, onehot, kc, nw, bs1, bs2, bn1, bn2, ea, be, ba, bih, bhh,
      Wp, bp, frags, adj, out);
}
